// TokenMergeModule_63350767616084
// MI455X (gfx1250) — compile-verified, matrix operand order checked by reading, not run on hardware
//
#include <hip/hip_runtime.h>
#include <hip/hip_bf16.h>

typedef __attribute__((ext_vector_type(16))) __bf16 v16bf;
typedef __attribute__((ext_vector_type(8)))  float  v8f;

#define BB    4
#define SS    4096
#define DD    768
#define GDIM  64
#define NORG  4096
#define RR    1024
#define SOUT  (SS - RR)      // 3072
#define KSTEPS (DD / 32)     // 24
#define NTILES (GDIM / 16)   // 4

// workspace byte offsets
#define WS_G    ((size_t)0)                               // float[B*S*64]  = 4 MB
#define WS_GN   (WS_G   + (size_t)BB*SS*GDIM*4)           // float[B*S]
#define WS_SIM  (WS_GN  + (size_t)BB*SS*4)                // float[B*S] (padded)
#define WS_SEL  (WS_SIM + (size_t)BB*SS*4)                // int[B*S]
#define WS_KEEP (WS_SEL + (size_t)BB*SS*4)                // int[B*SOUT]

// -------------------- Stage A: g = x @ W^T via bf16 WMMA --------------------
__global__ __launch_bounds__(256)
void k_proj_wmma(const float* __restrict__ x, const float* __restrict__ W,
                 float* __restrict__ g)
{
    extern __shared__ __bf16 wlds[]; // [KSTEPS][NTILES][32 lanes][16 bf16] = 96 KB
    const int tid = threadIdx.x;

    // Fill W fragments, pre-swizzled into the B-operand lane layout.
    for (int e = tid; e < KSTEPS * NTILES * 32 * 16; e += 256) {
        int j    = e & 15;
        int lane = (e >> 4) & 31;
        int nt   = (e >> 9) & 3;
        int kk   = e >> 11;
        int n    = nt * 16 + (lane & 15);
        int h    = lane >> 4;
        int base = j >> 1;
        int k    = ((base < 4) ? (2 * base) : (16 + 2 * (base - 4))) + (j & 1) + h * 8;
        wlds[e]  = (__bf16)W[n * DD + kk * 32 + k];
    }
    __syncthreads();

    const int wid  = tid >> 5;
    const int lane = tid & 31;
    const int h    = lane >> 4;
    const int ln   = lane & 15;
    const int rowBase = blockIdx.x * 128 + wid * 16;
    const int arow    = rowBase + ln;

    v8f acc[NTILES] = {};

    for (int kk = 0; kk < KSTEPS; ++kk) {
        v16bf av;
        #pragma unroll
        for (int p = 0; p < 8; ++p) {
            int dpair = ((p < 4) ? (2 * p) : (16 + 2 * (p - 4))) + h * 8;
            float2 v  = *(const float2*)(x + (size_t)arow * DD + kk * 32 + dpair);
            av[2 * p]     = (__bf16)v.x;
            av[2 * p + 1] = (__bf16)v.y;
        }
        #pragma unroll
        for (int nt = 0; nt < NTILES; ++nt) {
            v16bf bv = *(const v16bf*)(wlds + ((size_t)(kk * NTILES + nt) * 32 + lane) * 16);
            acc[nt] = __builtin_amdgcn_wmma_f32_16x16x32_bf16(
                          false, av, false, bv, (short)0, acc[nt], false, false);
        }
    }

    #pragma unroll
    for (int nt = 0; nt < NTILES; ++nt) {
        #pragma unroll
        for (int v = 0; v < 8; ++v) {
            int row = rowBase + v + 8 * h;
            g[(size_t)row * GDIM + nt * 16 + ln] = acc[nt][v];
        }
    }
}

// -------------------- Stage B: row norms (b128 loads) --------------------
__global__ void k_norm(const float* __restrict__ g, float* __restrict__ gn)
{
    int row = blockIdx.x * blockDim.x + threadIdx.x;
    if (row >= BB * SS) return;
    const float4* gr = (const float4*)(g + (size_t)row * GDIM);
    float s = 0.f;
    #pragma unroll
    for (int i = 0; i < GDIM / 4; ++i) {
        float4 v = gr[i];
        s += v.x * v.x + v.y * v.y + v.z * v.z + v.w * v.w;
    }
    gn[row] = sqrtf(s);
}

// ---------------- Stage C: adjacent cosine similarity (b128 loads) ----------------
__global__ void k_sim(const float* __restrict__ g, const float* __restrict__ gn,
                      float* __restrict__ sim)
{
    int idx = blockIdx.x * blockDim.x + threadIdx.x;
    if (idx >= BB * (SS - 1)) return;
    int b = idx / (SS - 1), s = idx % (SS - 1);
    int row = b * SS + s;
    const float4* g0 = (const float4*)(g + (size_t)row * GDIM);
    const float4* g1 = g0 + GDIM / 4;
    float d = 0.f;
    #pragma unroll
    for (int i = 0; i < GDIM / 4; ++i) {
        float4 a = g0[i], c = g1[i];
        d += a.x * c.x + a.y * c.y + a.z * c.z + a.w * c.w;
    }
    float n0 = fmaxf(gn[row], 1e-12f), n1 = fmaxf(gn[row + 1], 1e-12f);
    sim[b * SS + s] = d / (n0 * n1);
}

// ---------- Stage D: per-batch bitonic argsort + greedy disjoint pick ----------
__global__ __launch_bounds__(512)
void k_select(const float* __restrict__ sim, int* __restrict__ sel,
              int* __restrict__ keep)
{
    __shared__ float         skey[SS];
    __shared__ int           sidx[SS];
    __shared__ unsigned char used[SS];
    __shared__ unsigned char sll[SS];
    const int b   = blockIdx.x;
    const int tid = threadIdx.x;

    for (int i = tid; i < SS; i += 512) {
        skey[i] = (i < SS - 1) ? -sim[b * SS + i] : __builtin_inff();
        sidx[i] = i;
        used[i] = 0;
        sll[i]  = 0;
    }
    __syncthreads();

    // ascending bitonic sort on (key, idx) -> stable argsort(-sim)
    for (int k = 2; k <= SS; k <<= 1) {
        for (int j = k >> 1; j > 0; j >>= 1) {
            for (int i = tid; i < SS; i += 512) {
                int l = i ^ j;
                if (l > i) {
                    bool asc = ((i & k) == 0);
                    float ka = skey[i], kb = skey[l];
                    int   ia = sidx[i], ib = sidx[l];
                    bool greater = (ka > kb) || (ka == kb && ia > ib);
                    if (greater == asc) {
                        skey[i] = kb; skey[l] = ka;
                        sidx[i] = ib; sidx[l] = ia;
                    }
                }
            }
            __syncthreads();
        }
    }

    if (tid == 0) {
        int cnt = 0;
        for (int q = 0; q < 2 * (SS - 1); ++q) {
            int p = (q < SS - 1) ? sidx[q] : (q - (SS - 1));
            if (p >= SS - 1) continue;  // skip padded entry
            if (!used[p] && !used[p + 1] && cnt < RR) {
                used[p] = 1; used[p + 1] = 1; sll[p] = 1; ++cnt;
            }
        }
        int out = 0;
        for (int i = 0; i < SS && out < SOUT; ++i) {
            bool km = (i == 0) || !sll[i - 1];
            if (km) keep[b * SOUT + (out++)] = i;
        }
    }
    __syncthreads();
    for (int i = tid; i < SS; i += 512) sel[b * SS + i] = sll[i];
}

// ---------- Stage E: gather + merge (128-bit streaming + prefetch) ----------
__global__ __launch_bounds__(256)
void k_merge(const float* __restrict__ x, const float* __restrict__ src,
             const int* __restrict__ pos, const int* __restrict__ span,
             const float* __restrict__ gn, const int* __restrict__ sel,
             const int* __restrict__ keep,
             float* __restrict__ xo, float* __restrict__ so,
             float* __restrict__ po, float* __restrict__ spo)
{
    const int t   = blockIdx.x;       // 0 .. B*SOUT-1
    const int b   = t / SOUT;
    const int tid = threadIdx.x;
    const int i   = keep[t];
    const int row = b * SS + i;
    const int merged = (i < SS - 1) ? sel[row] : 0;

    float wi  = gn[row];
    float wj  = merged ? gn[row + 1] : 0.f;
    float inv = 1.f / (wi + wj + 1e-8f);

    const float4* xi = (const float4*)(x + (size_t)row * DD);
    const float4* xj = xi + DD / 4;
    float4* xop = (float4*)(xo + (size_t)t * DD);

    const float4* si = (const float4*)(src + (size_t)row * NORG);
    const float4* sj = si + NORG / 4;
    float4* sop = (float4*)(so + (size_t)t * NORG);

    // prefetch the merge-partner rows (data-dependent gather)
    if (merged) {
        __builtin_prefetch(xj + tid, 0, 1);
        __builtin_prefetch(sj + tid, 0, 1);
    }

    for (int c = tid; c < DD / 4; c += 256) {
        float4 a = xi[c], r;
        if (merged) {
            float4 bq = xj[c];
            r.x = (wi * a.x + wj * bq.x) * inv;
            r.y = (wi * a.y + wj * bq.y) * inv;
            r.z = (wi * a.z + wj * bq.z) * inv;
            r.w = (wi * a.w + wj * bq.w) * inv;
        } else {
            r = a;
        }
        xop[c] = r;
    }

    for (int c = tid; c < NORG / 4; c += 256) {
        float4 a = si[c], r;
        if (merged) {
            float4 bq = sj[c];
            r.x = a.x + bq.x; r.y = a.y + bq.y;
            r.z = a.z + bq.z; r.w = a.w + bq.w;
        } else {
            r = a;
        }
        sop[c] = r;
    }

    if (tid == 0) {
        po[t]  = (float)pos[row];
        spo[t] = (float)(merged ? (span[row] + span[row + 1]) : span[row]);
    }
}

extern "C" void kernel_launch(void* const* d_in, const int* in_sizes, int n_in,
                              void* d_out, int out_size, void* d_ws, size_t ws_size,
                              hipStream_t stream)
{
    const float* x    = (const float*)d_in[0];
    const float* src  = (const float*)d_in[1];
    const int*   pos  = (const int*)d_in[2];
    const int*   span = (const int*)d_in[3];
    const float* W    = (const float*)d_in[4];
    // d_in[5] = r (1024) baked in as RR

    char*  ws   = (char*)d_ws;
    float* g    = (float*)(ws + WS_G);
    float* gn   = (float*)(ws + WS_GN);
    float* sim  = (float*)(ws + WS_SIM);
    int*   sel  = (int*)(ws + WS_SEL);
    int*   keep = (int*)(ws + WS_KEEP);

    float* out = (float*)d_out;
    float* xo  = out;
    float* so  = xo + (size_t)BB * SOUT * DD;
    float* po  = so + (size_t)BB * SOUT * NORG;
    float* spo = po + (size_t)BB * SOUT;

    const size_t wbytes = (size_t)KSTEPS * NTILES * 32 * 16 * sizeof(__bf16); // 96 KB
    k_proj_wmma<<<(BB * SS) / 128, 256, wbytes, stream>>>(x, W, g);
    k_norm<<<(BB * SS + 255) / 256, 256, 0, stream>>>(g, gn);
    k_sim<<<(BB * (SS - 1) + 255) / 256, 256, 0, stream>>>(g, gn, sim);
    k_select<<<BB, 512, 0, stream>>>(sim, sel, keep);
    k_merge<<<BB * SOUT, 256, 0, stream>>>(x, src, pos, span, gn, sel, keep,
                                           xo, so, po, spo);
}
